// PTOutputOnlyMoE_51462298141173
// MI455X (gfx1250) — matrix operand order repaired and checked by reading, not compiled, not run on hardware
//
#include <hip/hip_runtime.h>
#include <hip/hip_bf16.h>
#include <math.h>

typedef __attribute__((ext_vector_type(16))) _Float16 v16h;
typedef __attribute__((ext_vector_type(8)))  _Float16 v8h;
typedef __attribute__((ext_vector_type(4)))  _Float16 v4h;
typedef __attribute__((ext_vector_type(8)))  float    v8f;
typedef __attribute__((ext_vector_type(4)))  int      v4i;

#define Bb 4
#define Tt 2048
#define Dd 1024
#define Hh 4096
#define Ee 8
#define Ss (Bb*Tt)   // 8192
#define Cc 1024      // capacity

union Frag16 { v16h v; v8h h[2]; };

// ---- async global->LDS path (guarded fallback to sync staging) ------------
#if defined(__gfx1250__) && __has_builtin(__builtin_amdgcn_global_load_async_to_lds_b128) && __has_builtin(__builtin_amdgcn_s_wait_asynccnt)
#define USE_ASYNC 1
#else
#define USE_ASYNC 0
#endif

#if USE_ASYNC
__device__ __forceinline__ void async_cp16(const _Float16* g, _Float16* l) {
  __builtin_amdgcn_global_load_async_to_lds_b128(
      (__attribute__((address_space(1))) v4i*)(g),
      (__attribute__((address_space(3))) v4i*)(l), 0, 0);
}
#endif

// ---------------------------------------------------------------- zero output
__global__ __launch_bounds__(256) void k_zero(float4* __restrict__ out) {
  out[(size_t)blockIdx.x * 256 + threadIdx.x] = make_float4(0.f, 0.f, 0.f, 0.f);
}

// ---------------------------------------------------------------- gating
// one wave (32 lanes) per token; logits = x[tok,:] @ wg (D x E), softmax+argmax
__global__ __launch_bounds__(256) void k_gate(const float* __restrict__ x,
                                              const float* __restrict__ wg,
                                              int* __restrict__ eidx,
                                              float* __restrict__ gate) {
  const int lane = threadIdx.x & 31;
  const int wid  = threadIdx.x >> 5;
  const int tok  = blockIdx.x * 8 + wid;
  const float* xr = x + (size_t)tok * Dd;

  float acc[Ee];
#pragma unroll
  for (int e = 0; e < Ee; ++e) acc[e] = 0.f;

  for (int d = lane; d < Dd; d += 32) {
    float xv = xr[d];
    const float4* wr = (const float4*)(wg + (size_t)d * Ee);
    float4 w0 = wr[0], w1 = wr[1];
    acc[0] += xv * w0.x; acc[1] += xv * w0.y; acc[2] += xv * w0.z; acc[3] += xv * w0.w;
    acc[4] += xv * w1.x; acc[5] += xv * w1.y; acc[6] += xv * w1.z; acc[7] += xv * w1.w;
  }
#pragma unroll
  for (int e = 0; e < Ee; ++e) {
#pragma unroll
    for (int off = 16; off > 0; off >>= 1)
      acc[e] += __shfl_xor(acc[e], off, 32);
  }
  if (lane == 0) {
    int best = 0; float mx = acc[0];
#pragma unroll
    for (int e = 1; e < Ee; ++e) { if (acc[e] > mx) { mx = acc[e]; best = e; } }
    float sum = 0.f;
#pragma unroll
    for (int e = 0; e < Ee; ++e) sum += expf(acc[e] - mx);
    eidx[tok] = best;
    gate[tok] = 1.0f / sum;      // softmax value at argmax = exp(0)/sum
  }
}

// ---------------------------------------------------------------- slot assign
// single block; exact in-order cumsum per expert; builds inverse dispatch table
__global__ __launch_bounds__(256) void k_slot(const int* __restrict__ eidx,
                                              float* __restrict__ gate,
                                              int* __restrict__ dtok) {
  __shared__ int cnt[256][Ee];
  const int tid = threadIdx.x;
  for (int i = tid; i < Ee * Cc; i += 256) dtok[i] = -1;

  int local[Ee];
#pragma unroll
  for (int e = 0; e < Ee; ++e) local[e] = 0;
  const int s0 = tid * (Ss / 256);        // 32 tokens per thread, in token order
  for (int i = 0; i < Ss / 256; ++i) local[eidx[s0 + i]]++;
#pragma unroll
  for (int e = 0; e < Ee; ++e) cnt[tid][e] = local[e];
  __syncthreads();
  if (tid < Ee) {                          // exclusive scan over 256 chunks
    int run = 0;
    for (int t = 0; t < 256; ++t) { int c = cnt[t][tid]; cnt[t][tid] = run; run += c; }
  }
  __syncthreads();
  int base[Ee];
#pragma unroll
  for (int e = 0; e < Ee; ++e) base[e] = cnt[tid][e];
  for (int i = 0; i < Ss / 256; ++i) {
    const int s = s0 + i;
    const int e = eidx[s];
    const int sl = base[e]++;
    if (sl < Cc) dtok[e * Cc + sl] = s;
    else         gate[s] = 0.f;            // dropped token
  }
}

// ---------------------------------------------------------------- f32 -> f16
__global__ __launch_bounds__(256) void k_cvt(const float4* __restrict__ in,
                                             v4h* __restrict__ out) {
  const size_t i = (size_t)blockIdx.x * 256 + threadIdx.x;
  const float4 v = in[i];
  v4h o;
  o.x = (_Float16)v.x; o.y = (_Float16)v.y; o.z = (_Float16)v.z; o.w = (_Float16)v.w;
  out[i] = o;
}

// ---------------------------------------------------------------- dispatch
__global__ __launch_bounds__(256) void k_disp(const float* __restrict__ x,
                                              const int* __restrict__ dtok,
                                              v4h* __restrict__ disp) {
  const int row = blockIdx.x;             // e*Cc + c
  const int tok = dtok[row];
  const int d0  = threadIdx.x * 4;
  v4h o;
  if (tok >= 0) {
    const float4 v = *(const float4*)(x + (size_t)tok * Dd + d0);
    o.x = (_Float16)v.x; o.y = (_Float16)v.y; o.z = (_Float16)v.z; o.w = (_Float16)v.w;
  } else {
    o.x = (_Float16)0.f; o.y = (_Float16)0.f; o.z = (_Float16)0.f; o.w = (_Float16)0.f;
  }
  disp[((size_t)row * Dd + d0) >> 2] = o;
}

// ---------------------------------------------------------------- WMMA GEMM
// block tile 128 x 128, 8 waves (2x4), wave tile 64 x 32 (4x2 frags), K-step 32.
// Async path: double-buffered LDS fed by GLOBAL_LOAD_ASYNC_TO_LDS_B128, with
// explicit buffer-pointer swap + unroll(disable) so accumulators stay in place.
// RELU16: out = f16(relu(acc)) -> O16[e,m,n] ; else scatter acc*gate -> Yout.
template<int KTOT, int NTOT, bool RELU16>
__global__ __launch_bounds__(256) void k_gemm(const _Float16* __restrict__ A,
                                              const _Float16* __restrict__ Bm,
                                              _Float16* __restrict__ O16,
                                              float* __restrict__ Yout,
                                              const int* __restrict__ dtok,
                                              const float* __restrict__ gate) {
  constexpr int FN   = 2;           // B frags per wave (wave tile N = 32)
  constexpr int BSTR = 128 + 8;     // Bs row stride in halfs (keeps 16B alignment)
  constexpr int NT   = KTOT / 32;

  const int tid  = threadIdx.x;
  const int lane = tid & 31;
  const int wid  = tid >> 5;
  const int e    = blockIdx.z;
  const int n0   = blockIdx.x * 128;
  const int m0   = blockIdx.y * 128;
  const int waveM = (wid >> 2) * 64;
  const int waveN = (wid & 3) * 32;

  const _Float16* Ae = A  + (size_t)e * Cc * KTOT + (size_t)m0 * KTOT;
  const _Float16* Be = Bm + (size_t)e * KTOT * NTOT + n0;

  v8f acc[4][FN];
#pragma unroll
  for (int i = 0; i < 4; ++i)
#pragma unroll
    for (int j = 0; j < FN; ++j)
      acc[i][j] = (v8f){0.f,0.f,0.f,0.f,0.f,0.f,0.f,0.f};

  const int arow = tid >> 2;          // 0..63
  const int acg  = (tid & 3) * 8;     // 0,8,16,24 halfs
  const int brow = tid >> 3;          // 0..31
  const int bcg  = (tid & 7) * 8;     // 0..56 halfs; +64*j covers 128 cols
  const int mh8  = (lane >> 4) * 8;   // A frag K-chunk select per lane half
  const int ml   = lane & 15;

  // fragment compute from a staged (As,Bs) tile pair
  auto compute = [&](const _Float16 (*Asb)[40], const _Float16 (*Bsb)[BSTR]) {
    Frag16 bf[FN];
#pragma unroll
    for (int fn = 0; fn < FN; ++fn) {
      // 16-bit B 32x16 layout: lane%16 = N col, lane half selects K chunks {mh8.., 16+mh8..}
      const _Float16* br = &Bsb[0][waveN + fn * 16 + ml];
#pragma unroll
      for (int s = 0; s < 8; ++s) {
        bf[fn].v[s]     = br[(mh8 + s) * BSTR];
        bf[fn].v[8 + s] = br[(16 + mh8 + s) * BSTR];
      }
    }
#pragma unroll
    for (int fm = 0; fm < 4; ++fm) {
      // 16-bit A 16x32 layout: lane half selects K chunks {mh8.., 16+mh8..}
      Frag16 a;
      const _Float16* ar = &Asb[waveM + fm * 16 + ml][0];
      a.h[0] = *(const v8h*)(ar + mh8);
      a.h[1] = *(const v8h*)(ar + 16 + mh8);
#pragma unroll
      for (int fn = 0; fn < FN; ++fn)
        acc[fm][fn] = __builtin_amdgcn_wmma_f32_16x16x32_f16(
            false, a.v, false, bf[fn].v, (short)0, acc[fm][fn], false, false);
    }
  };

#if USE_ASYNC
  __shared__ alignas(16) _Float16 As[2][128][40];
  __shared__ alignas(16) _Float16 Bs[2][32][BSTR];

  auto stage = [&](int k0, _Float16 (*Asb)[40], _Float16 (*Bsb)[BSTR]) {
    async_cp16(Ae + (size_t)arow * KTOT + k0 + acg,        &Asb[arow][acg]);
    async_cp16(Ae + (size_t)(arow + 64) * KTOT + k0 + acg, &Asb[arow + 64][acg]);
#pragma unroll
    for (int j = 0; j < FN; ++j)
      async_cp16(Be + (size_t)(k0 + brow) * NTOT + bcg + 64 * j,
                 &Bsb[brow][bcg + 64 * j]);
  };

  _Float16 (*Acur)[40]   = As[0], (*Anxt)[40]   = As[1];
  _Float16 (*Bcur)[BSTR] = Bs[0], (*Bnxt)[BSTR] = Bs[1];

  stage(0, Acur, Bcur);
#pragma clang loop unroll(disable)
  for (int kt = 0; kt < NT; ++kt) {
    __builtin_amdgcn_s_wait_asynccnt(0);   // my slice of tile kt is in LDS
    __syncthreads();                       // everyone's slice is in LDS
    if (kt + 1 < NT)                       // prefetch next tile into other buffer
      stage((kt + 1) * 32, Anxt, Bnxt);
    compute(Acur, Bcur);
    { _Float16 (*t)[40]   = Acur; Acur = Anxt; Anxt = t; }
    { _Float16 (*t)[BSTR] = Bcur; Bcur = Bnxt; Bnxt = t; }
  }
#else
  __shared__ alignas(16) _Float16 As1[128][40];
  __shared__ alignas(16) _Float16 Bs1[32][BSTR];

  for (int kt = 0; kt < NT; ++kt) {
    const int k0 = kt * 32;
    *(v8h*)&As1[arow][acg]      = *(const v8h*)(Ae + (size_t)arow * KTOT + k0 + acg);
    *(v8h*)&As1[arow + 64][acg] = *(const v8h*)(Ae + (size_t)(arow + 64) * KTOT + k0 + acg);
#pragma unroll
    for (int j = 0; j < FN; ++j)
      *(v8h*)&Bs1[brow][bcg + 64 * j] =
          *(const v8h*)(Be + (size_t)(k0 + brow) * NTOT + bcg + 64 * j);
    if (k0 + 32 < KTOT) {
      __builtin_prefetch(Ae + (size_t)arow * KTOT + k0 + 32 + acg, 0, 1);
      __builtin_prefetch(Be + (size_t)(k0 + 32 + brow) * NTOT + bcg, 0, 1);
    }
    __syncthreads();
    compute(As1, Bs1);
    __syncthreads();
  }
#endif

  // epilogue (C/D layout: VGPR r, lanes0-15 -> M=r,N=lane; lanes16-31 -> M=8+r)
  const int nl = lane & 15;
  const int mh = (lane >> 4) * 8;
#pragma unroll
  for (int fm = 0; fm < 4; ++fm) {
#pragma unroll
    for (int r = 0; r < 8; ++r) {
      const int m = m0 + waveM + fm * 16 + mh + r;
      if (RELU16) {
#pragma unroll
        for (int fn = 0; fn < FN; ++fn) {
          float v = acc[fm][fn][r];
          v = v > 0.f ? v : 0.f;
          O16[(size_t)e * Cc * NTOT + (size_t)m * NTOT + n0 + waveN + fn * 16 + nl] =
              (_Float16)v;
        }
      } else {
        const int tok = dtok[e * Cc + m];
        if (tok >= 0) {
          const float g = gate[tok];
#pragma unroll
          for (int fn = 0; fn < FN; ++fn)
            Yout[(size_t)tok * Dd + n0 + waveN + fn * 16 + nl] = acc[fm][fn][r] * g;
        }
      }
    }
  }
}

// ---------------------------------------------------------------- launch
extern "C" void kernel_launch(void* const* d_in, const int* in_sizes, int n_in,
                              void* d_out, int out_size, void* d_ws, size_t ws_size,
                              hipStream_t stream) {
  (void)in_sizes; (void)n_in; (void)out_size; (void)ws_size;
  const float* x  = (const float*)d_in[0];   // [S, D]
  const float* wg = (const float*)d_in[1];   // [D, E]
  const float* w1 = (const float*)d_in[2];   // [E, D, H]
  const float* w2 = (const float*)d_in[3];   // [E, H, D]
  float* y = (float*)d_out;                  // [S, D]

  char* ws = (char*)d_ws;
  size_t off = 0;
  auto take = [&](size_t bytes) -> void* {
    void* p = ws + off;
    off += bytes;
    off = (off + 255) & ~(size_t)255;
    return p;
  };
  int*      eidx = (int*)take((size_t)Ss * 4);
  float*    gate = (float*)take((size_t)Ss * 4);
  int*      dtok = (int*)take((size_t)Ee * Cc * 4);
  _Float16* disp = (_Float16*)take((size_t)Ee * Cc * Dd * 2);   // 16 MB
  _Float16* w1h  = (_Float16*)take((size_t)Ee * Dd * Hh * 2);   // 64 MB
  _Float16* w2h  = (_Float16*)take((size_t)Ee * Hh * Dd * 2);   // 64 MB
  _Float16* hh   = (_Float16*)take((size_t)Ee * Cc * Hh * 2);   // 64 MB

  k_zero<<<(Ss * Dd) / (256 * 4), 256, 0, stream>>>((float4*)y);
  k_gate<<<Ss / 8, 256, 0, stream>>>(x, wg, eidx, gate);
  k_slot<<<1, 256, 0, stream>>>(eidx, gate, dtok);
  k_cvt<<<((size_t)Ee * Dd * Hh) / (256 * 4), 256, 0, stream>>>((const float4*)w1, (v4h*)w1h);
  k_cvt<<<((size_t)Ee * Hh * Dd) / (256 * 4), 256, 0, stream>>>((const float4*)w2, (v4h*)w2h);
  k_disp<<<Ee * Cc, 256, 0, stream>>>(x, dtok, (v4h*)disp);

  // GEMM1: h = relu(disp[E,C,D] @ w1h[E,D,H]) -> f16 hh[E,C,H]
  k_gemm<Dd, Hh, true><<<dim3(Hh / 128, Cc / 128, Ee), 256, 0, stream>>>(
      disp, w1h, hh, nullptr, nullptr, nullptr);
  // GEMM2: y[tok] = (hh[E,C,H] @ w2h[E,H,D]) * gate[tok]
  k_gemm<Hh, Dd, false><<<dim3(Dd / 128, Cc / 128, Ee), 256, 0, stream>>>(
      hh, w2h, nullptr, y, dtok, gate);
}
